// LSTM_49684181680146
// MI455X (gfx1250) — matrix operand order repaired and checked by reading, not compiled, not run on hardware
//
#include <hip/hip_runtime.h>

// Bidirectional LSTM (H=39, B=512, T=2048), fused persistent scan.
// grid = (32 batch tiles, 2 directions), block = 64 threads (2 waves).
// Each wave owns 5 of the 10 gate N-tiles; the 15 f16 WMMA B-tiles per wave
// (16x96 x 96x80 slice of [W_ih;W_hh]) are held in VGPRs for the whole scan.
// Per step: A=[x_t | h_{t-1}] (16x96 f16) staged in LDS, 15 v_wmma per wave,
// gates spilled transposed to LDS, elementwise cell update, h fed back.

constexpr int kH  = 39;
constexpr int kB  = 512;
constexpr int kT  = 2048;
constexpr int kXHW = 96;   // A staging width: [x:0..38 | pad | h:48..86 | pad]

typedef __attribute__((ext_vector_type(16))) _Float16 v16h;
typedef __attribute__((ext_vector_type(8)))  _Float16 v8h;
typedef __attribute__((ext_vector_type(8)))  float    v8f;
typedef __attribute__((ext_vector_type(4)))  float    v4f;

__device__ __forceinline__ float fast_sigmoid(float x) {
    return 1.0f / (1.0f + __expf(-x));
}
__device__ __forceinline__ float fast_tanh(float x) {
    return 2.0f / (1.0f + __expf(-2.0f * x)) - 1.0f;
}

__global__ __launch_bounds__(64)
void lstm_bidir_wmma(const float* __restrict__ x,
                     const float* __restrict__ Wih_f, const float* __restrict__ Whh_f,
                     const float* __restrict__ bih_f, const float* __restrict__ bhh_f,
                     const float* __restrict__ Wih_b, const float* __restrict__ Whh_b,
                     const float* __restrict__ bih_b, const float* __restrict__ bhh_b,
                     float* __restrict__ out)
{
    __shared__ __align__(16) _Float16 xh[16 * kXHW];   // A staging [m][col]
    __shared__ __align__(16) float    gT[160 * 16];    // gates [n][m]

    const int tid  = threadIdx.x;        // 0..63
    const int lane = tid & 31;
    const int wv   = tid >> 5;           // wave id: N-tiles wv*5 .. wv*5+4
    const int m16  = lane & 15;
    const int hs   = (lane >> 4) & 1;    // half-wave selector
    const int b0   = blockIdx.x * 16;
    const int dir  = blockIdx.y;

    const float* Wih = dir ? Wih_b : Wih_f;
    const float* Whh = dir ? Whh_b : Whh_f;
    const float* bih = dir ? bih_b : bih_f;
    const float* bhh = dir ? bhh_b : bhh_f;

    // ---- one-time: this wave's 15 WMMA B-tiles -> VGPRs ----
    // element e of lane l, tile (nt,kc): N = nt*16 + (l&15), K = kc*32 + (l>>4)*16 + e
    v16h bmat[15];
    #pragma unroll
    for (int nt2 = 0; nt2 < 5; ++nt2) {
        const int n = (wv * 5 + nt2) * 16 + m16;
        #pragma unroll
        for (int kc = 0; kc < 3; ++kc) {
            v16h bm;
            #pragma unroll
            for (int e = 0; e < 16; ++e) {
                const int k = kc * 32 + hs * 16 + e;
                float w = 0.0f;
                if (n < 4 * kH) {
                    if (k < 48) { if (k < kH) w = Wih[n * kH + k]; }
                    else        { const int k2 = k - 48; if (k2 < kH) w = Whh[n * kH + k2]; }
                }
                bm[e] = (_Float16)w;
            }
            bmat[nt2 * 3 + kc] = bm;
        }
    }

    float biasv[5];
    #pragma unroll
    for (int nt2 = 0; nt2 < 5; ++nt2) {
        const int n = (wv * 5 + nt2) * 16 + m16;
        biasv[nt2] = (n < 4 * kH) ? (bih[n] + bhh[n]) : 0.0f;
    }

    for (int i = tid; i < 16 * kXHW; i += 64) xh[i] = (_Float16)0.0f;  // pads + h0

    // per-thread slots: fixed row m, contiguous j = jb..jb+9 (j==39 is pad)
    const int m  = tid >> 2;             // 0..15
    const int jb = (tid & 3) * 10;       // 0,10,20,30
    const size_t xrow = (size_t)(b0 + m) * (kT * kH) + jb;             // + t_in*kH + q
    const size_t orow = (size_t)(b0 + m) * (kT * 2 * kH) + dir * kH + jb; // + s*2*kH + q

    float creg[10];
    #pragma unroll
    for (int q = 0; q < 10; ++q) creg[q] = 0.0f;

    __syncthreads();

    for (int s = 0; s < kT; ++s) {
        const int t_in = dir ? (kT - 1 - s) : s;
        const size_t xbase = xrow + (size_t)t_in * kH;

        // stage x_t tile into xh cols 0..38 (10 contiguous elements per thread)
        #pragma unroll
        for (int q = 0; q < 10; ++q) {
            const int j = jb + q;
            const float xv = (j < kH) ? x[xbase + q] : 0.0f;
            xh[m * kXHW + j] = (_Float16)xv;
        }
        if (s + 1 < kT) {  // prefetch next timestep row (global_prefetch_b8)
            const int t_nx = dir ? (kT - 2 - s) : (s + 1);
            __builtin_prefetch(&x[xrow + (size_t)t_nx * kH], 0, 1);
        }
        __syncthreads();

        // A: three 16x32 f16 chunks from LDS (two b128 each)
        v16h a[3];
        #pragma unroll
        for (int kc = 0; kc < 3; ++kc) {
            const int base0 = m16 * kXHW + kc * 32 + hs * 16;
            const v8h lo = *(const v8h*)&xh[base0];
            const v8h hi = *(const v8h*)&xh[base0 + 8];
            #pragma unroll
            for (int e = 0; e < 8; ++e) { a[kc][e] = lo[e]; a[kc][8 + e] = hi[e]; }
        }

        // this wave's 5 gate N-tiles: 15 WMMAs, B resident in VGPRs
        #pragma unroll
        for (int nt2 = 0; nt2 < 5; ++nt2) {
            v8f acc;
            #pragma unroll
            for (int r = 0; r < 8; ++r) acc[r] = biasv[nt2];
            #pragma unroll
            for (int kc = 0; kc < 3; ++kc) {
                acc = __builtin_amdgcn_wmma_f32_16x16x32_f16(
                        false, a[kc], false, bmat[nt2 * 3 + kc],
                        (short)0, acc, false, false);
            }
            // spill D to gT[n][m]; this lane's rows are hs*8 .. hs*8+7
            float* gp = &gT[((wv * 5 + nt2) * 16 + m16) * 16 + hs * 8];
            v4f lo4, hi4;
            #pragma unroll
            for (int r = 0; r < 4; ++r) { lo4[r] = acc[r]; hi4[r] = acc[4 + r]; }
            *(v4f*)gp       = lo4;
            *(v4f*)(gp + 4) = hi4;
        }
        __syncthreads();

        // elementwise LSTM cell update (gate rows: i=0.., f=39.., g=78.., o=117..)
        const size_t obase = orow + (size_t)s * (2 * kH);
        #pragma unroll
        for (int q = 0; q < 10; ++q) {
            const int j = jb + q;
            const float gi = fast_sigmoid(gT[(j)          * 16 + m]);
            const float gf = fast_sigmoid(gT[(kH + j)     * 16 + m]);
            const float gg = fast_tanh   (gT[(2 * kH + j) * 16 + m]);
            const float go = fast_sigmoid(gT[(3 * kH + j) * 16 + m]);
            const float c  = gf * creg[q] + gi * gg;
            creg[q] = c;
            const float h = go * fast_tanh(c);
            xh[m * kXHW + 48 + j] = (_Float16)h;   // feed back (col 87 pad harmless)
            if (j < kH) out[obase + q] = h;
        }
        __syncthreads();
    }
}

extern "C" void kernel_launch(void* const* d_in, const int* in_sizes, int n_in,
                              void* d_out, int out_size, void* d_ws, size_t ws_size,
                              hipStream_t stream) {
    (void)in_sizes; (void)n_in; (void)out_size; (void)d_ws; (void)ws_size;
    const float* x     = (const float*)d_in[0];
    const float* Wih_f = (const float*)d_in[1];
    const float* Whh_f = (const float*)d_in[2];
    const float* bih_f = (const float*)d_in[3];
    const float* bhh_f = (const float*)d_in[4];
    const float* Wih_b = (const float*)d_in[5];
    const float* Whh_b = (const float*)d_in[6];
    const float* bih_b = (const float*)d_in[7];
    const float* bhh_b = (const float*)d_in[8];
    float* out = (float*)d_out;

    dim3 grid(kB / 16, 2);
    dim3 block(64);
    lstm_bidir_wmma<<<grid, block, 0, stream>>>(x, Wih_f, Whh_f, bih_f, bhh_f,
                                                Wih_b, Whh_b, bih_b, bhh_b, out);
}
